// HAN_71107478553205
// MI455X (gfx1250) — matrix operand order repaired and checked by reading, not compiled, not run on hardware
//
#include <hip/hip_runtime.h>
#include <hip/hip_bf16.h>
#include <math.h>

// ---------------------------------------------------------------------------
// HAN (2 layers) on MI455X / gfx1250.
//
// Roofline: dense GEMMs total < 10 GFLOP (negligible on the WMMA pipes);
// the edge gather/scatter (4M edges x 2 layers x ~512B) dominates and is
// memory/atomic bound (~GBs of traffic vs 23.3 TB/s HBM; 77MB node tables
// largely live in the 192MB L2). Therefore:
//   * dense math -> v_wmma_f32_16x16x32_bf16, one wave per 16x16 tile,
//     K-loop fully unrolled via templates; B operands pre-packed to bf16
//     WMMA-fragment layout in LDS (2x ds_load_b128 per operand, no strided
//     scalar LDS reads, no per-iteration cvt for B).
//   * edge softmax/scatter -> f32 VALU + native global_atomic_add_f32,
//     alpha recomputed per pass instead of materializing 128MB of weights.
// ---------------------------------------------------------------------------

typedef __attribute__((ext_vector_type(16))) __bf16 v16bf;
typedef __attribute__((ext_vector_type(8)))  float  v8f;

#define TPB 256

__device__ __forceinline__ float leaky02(float x) { return x > 0.f ? x : 0.2f * x; }

__device__ __forceinline__ void atomAddF(float* p, float v) {
  // lowers to global_atomic_add_f32 (no CAS loop)
  __hip_atomic_fetch_add(p, v, __ATOMIC_RELAXED, __HIP_MEMORY_SCOPE_AGENT);
}

__device__ __forceinline__ void atomMaxF(float* p, float v) {
  // classic monotone-bits trick; init value must be -INF
  if (v >= 0.f) atomicMax((int*)p, __float_as_int(v));
  else          atomicMin((unsigned int*)p, __float_as_uint(v));
}

// ---------------------------------------------------------------------------
// Templated WMMA GEMM:  D[N,NOUT] = X[N,K] @ W[K,NOUT] + bias
// TANH=false: store D.      TANH=true: acc[col] += sum_rows tanh(D[row,col]).
// One wave per 16x16 tile. N, NOUT multiples of 16; K in {32,64}.
//
// B fragments are pre-packed in LDS in the exact 16-bit-B WMMA layout
// (B 32x16: lanes0-15 hold K=kb..kb+7,kb+16..kb+23, lanes16-31 hold K=kb+8..kb+15,kb+24..kb+31,
//  lane&15 = N column), so each wave loads its B operand as 32 contiguous
// bytes per lane (2x ds_load_b128).
// ---------------------------------------------------------------------------
template <int K, int NOUT, bool TANH>
__global__ void k_gemm_t(const float* __restrict__ X, const float* __restrict__ W,
                         const float* __restrict__ bias, float* __restrict__ OutOrAcc,
                         int N)
{
  constexpr int KSTEPS = K / 32;
  constexpr int TILESN = NOUT / 16;
  constexpr int NFRAG  = TILESN * KSTEPS * 32;     // fragment-slices (16 bf16 each)
  __shared__ __align__(32) __bf16 sB[NFRAG * 16];  // <= 8 KB

  // cooperative pack: slice idx -> (tn, kbi, lane)
  for (int idx = threadIdx.x; idx < NFRAG; idx += TPB) {
    const int pl    = idx & 31;
    const int kbi   = (idx >> 5) % KSTEPS;
    const int tn    = (idx >> 5) / KSTEPS;
    const int col   = tn * 16 + (pl & 15);
    const int kb2   = kbi * 32 + (pl >> 4) * 8;
    __bf16* dstp = &sB[idx * 16];
#pragma unroll
    for (int e = 0; e < 16; ++e) dstp[e] = (__bf16)W[(kb2 + (e < 8 ? e : e + 8)) * NOUT + col];
  }
  __syncthreads();

  const int  lane     = threadIdx.x & 31;
  const int  wave     = threadIdx.x >> 5;
  const long tilesTot = (long)(N >> 4) * TILESN;
  const long tile     = (long)blockIdx.x * (TPB / 32) + wave;
  if (tile >= tilesTot) return;                    // wave-uniform: EXEC stays all-1s
  const int tm = (int)(tile / TILESN);
  const int tn = (int)(tile % TILESN);

  const int l15   = lane & 15;
  const int khalf = lane >> 4;
  const int row   = tm * 16 + l15;                 // A-matrix M row
  const int col   = tn * 16 + l15;                 // B/C/D N column

  v8f c;
  const float bv = bias[col];
#pragma unroll
  for (int r = 0; r < 8; ++r) c[r] = bv;           // fold bias into accumulator

#pragma unroll
  for (int kbi = 0; kbi < KSTEPS; ++kbi) {
    // A 16x32 bf16 layout: elems 0..7 -> K=kb+khalf*8+e, elems 8..15 -> +16
    const float* xr = X + (long)row * K + kbi * 32 + khalf * 8;
    v16bf a;
#pragma unroll
    for (int e = 0; e < 8; ++e) {
      a[e]     = (__bf16)xr[e];
      a[e + 8] = (__bf16)xr[16 + e];
    }
    const v16bf b = *(const v16bf*)&sB[((tn * KSTEPS + kbi) * 32 + lane) * 16];
    c = __builtin_amdgcn_wmma_f32_16x16x32_bf16(false, a, false, b, (short)0, c,
                                                false, false);
  }

  if constexpr (TANH) {
    float s = 0.f;
#pragma unroll
    for (int r = 0; r < 8; ++r) s += tanhf(c[r]);
    atomAddF(&OutOrAcc[col], s);                   // column-sum over all rows
  } else {
#pragma unroll
    for (int r = 0; r < 8; ++r)                    // C/D: M = r + 8*khalf
      OutOrAcc[(long)(tm * 16 + khalf * 8 + r) * NOUT + col] = c[r];
  }
}

// ---------------------------------------------------------------------------
// per-node attention logits: al[n,h] = sum_d H[n,h*D+d] * a[h,d]
// ---------------------------------------------------------------------------
__global__ void k_node_scores(const float* __restrict__ Hf, const float* __restrict__ av,
                              float* __restrict__ al, long N, int H, int D)
{
  long i = (long)blockIdx.x * TPB + threadIdx.x;
  if (i >= N * H) return;
  int  h = (int)(i % H);
  long n = i / H;
  const float* x = Hf + (n * H + h) * D;
  const float* a = av + (long)h * D;
  float s = 0.f;
  for (int d = 0; d < D; ++d) s += x[d] * a[d];
  al[i] = s;
}

__global__ void k_fill(float* __restrict__ p, float v, long n)
{
  long i = (long)blockIdx.x * TPB + threadIdx.x;
  if (i < n) p[i] = v;
}

__global__ void k_relu(float* __restrict__ p, long n)
{
  long i = (long)blockIdx.x * TPB + threadIdx.x;
  if (i < n) p[i] = fmaxf(p[i], 0.f);
}

// ---------------------------------------------------------------------------
// segment-softmax over destinations, three passes (recompute alpha each pass)
// ---------------------------------------------------------------------------
__global__ void k_edge_max(const float* __restrict__ alS, const float* __restrict__ alD,
                           const int* __restrict__ src, const int* __restrict__ dst,
                           float* __restrict__ amax, long E, int H)
{
  long i = (long)blockIdx.x * TPB + threadIdx.x;
  if (i >= E * H) return;
  long e = i / H; int h = (int)(i % H);
  long s = src[e], d = dst[e];
  float a = leaky02(alS[s * H + h] + alD[d * H + h]);
  atomMaxF(&amax[d * H + h], a);
}

__global__ void k_edge_den(const float* __restrict__ alS, const float* __restrict__ alD,
                           const int* __restrict__ src, const int* __restrict__ dst,
                           const float* __restrict__ amax, float* __restrict__ den,
                           long E, int H)
{
  long i = (long)blockIdx.x * TPB + threadIdx.x;
  if (i >= E * H) return;
  long e = i / H; int h = (int)(i % H);
  long s = src[e], d = dst[e];
  float a = leaky02(alS[s * H + h] + alD[d * H + h]);
  atomAddF(&den[d * H + h], __expf(a - amax[d * H + h]));
}

__global__ void k_edge_msg(const float* __restrict__ hs,
                           const float* __restrict__ alS, const float* __restrict__ alD,
                           const float* __restrict__ amax, const float* __restrict__ den,
                           const int* __restrict__ src, const int* __restrict__ dst,
                           float* __restrict__ out, long E, int H, int D)
{
  long i = (long)blockIdx.x * TPB + threadIdx.x;
  if (i >= E * H) return;
  long e = i / H; int h = (int)(i % H);
  long s = src[e], d = dst[e];
  float a = leaky02(alS[s * H + h] + alD[d * H + h]);
  float w = __expf(a - amax[d * H + h]) / (den[d * H + h] + 1e-16f);
  const float* xs = hs  + (s * H + h) * D;
  float*       op = out + (d * H + h) * D;
  for (int k = 0; k < D; ++k) atomAddF(&op[k], xs[k] * w);
}

// ---------------------------------------------------------------------------
// semantic attention (T=2): score_t = q . (acc_t / N); attn = softmax
// ---------------------------------------------------------------------------
__global__ void k_semantic(const float* __restrict__ acc0, const float* __restrict__ acc1,
                           const float* __restrict__ q, float invN, int F,
                           float* __restrict__ attn)
{
  if (threadIdx.x == 0 && blockIdx.x == 0) {
    float s0 = 0.f, s1 = 0.f;
    for (int f = 0; f < F; ++f) { s0 += q[f] * acc0[f] * invN; s1 += q[f] * acc1[f] * invN; }
    float m  = fmaxf(s0, s1);
    float e0 = __expf(s0 - m), e1 = __expf(s1 - m);
    float inv = 1.f / (e0 + e1);
    attn[0] = e0 * inv; attn[1] = e1 * inv;
  }
}

__global__ void k_combine(const float* __restrict__ o0, const float* __restrict__ o1,
                          const float* __restrict__ attn, float* __restrict__ out, long n)
{
  long i = (long)blockIdx.x * TPB + threadIdx.x;
  if (i < n) out[i] = attn[0] * o0[i] + attn[1] * o1[i];
}

// ---------------------------------------------------------------------------
// host-side orchestration
// ---------------------------------------------------------------------------
static inline unsigned cdiv(long a, long b) { return (unsigned)((a + b - 1) / b); }

static void edge_out(const float* hs, const float* hd,
                     const float* a_s, const float* a_d,
                     const int* src, const int* dst, long E,
                     long Ns, long Nd, int H, int D,
                     float* alS, float* alD, float* amax, float* den,
                     float* out, hipStream_t st)
{
  const int  F  = H * D;
  const long EH = E * H;
  k_node_scores<<<cdiv(Ns * H, TPB), TPB, 0, st>>>(hs, a_s, alS, Ns, H, D);
  k_node_scores<<<cdiv(Nd * H, TPB), TPB, 0, st>>>(hd, a_d, alD, Nd, H, D);
  k_fill<<<cdiv(Nd * H, TPB), TPB, 0, st>>>(amax, -INFINITY, Nd * H);
  k_fill<<<cdiv(Nd * H, TPB), TPB, 0, st>>>(den, 0.f, Nd * H);
  k_fill<<<cdiv(Nd * F, TPB), TPB, 0, st>>>(out, 0.f, Nd * F);
  k_edge_max<<<cdiv(EH, TPB), TPB, 0, st>>>(alS, alD, src, dst, amax, E, H);
  k_edge_den<<<cdiv(EH, TPB), TPB, 0, st>>>(alS, alD, src, dst, amax, den, E, H);
  k_edge_msg<<<cdiv(EH, TPB), TPB, 0, st>>>(hs, alS, alD, amax, den, src, dst, out, E, H, D);
  k_relu<<<cdiv(Nd * F, TPB), TPB, 0, st>>>(out, Nd * F);
}

static void gemm(const float* X, const float* W, const float* b, float* Out,
                 long N, int K, int Nout, hipStream_t st)
{
  long tiles = (N >> 4) * (long)(Nout >> 4);
  unsigned blocks = cdiv(tiles, TPB / 32);
  if (K == 64 && Nout == 64)
    k_gemm_t<64, 64, false><<<blocks, TPB, 0, st>>>(X, W, b, Out, (int)N);
  else if (K == 64 && Nout == 32)
    k_gemm_t<64, 32, false><<<blocks, TPB, 0, st>>>(X, W, b, Out, (int)N);
}

static void tanh_acc(const float* X, const float* W, const float* b, float* acc,
                     long N, int F, hipStream_t st)
{
  long tiles = (N >> 4) * (long)(F >> 4);
  unsigned blocks = cdiv(tiles, TPB / 32);
  if (F == 64)
    k_gemm_t<64, 64, true><<<blocks, TPB, 0, st>>>(X, W, b, acc, (int)N);
  else
    k_gemm_t<32, 32, true><<<blocks, TPB, 0, st>>>(X, W, b, acc, (int)N);
}

static void group2(const float* o0, const float* o1, const float* kW, const float* kb,
                   const float* q, long N, int F, float* acc, float* attn, float* outp,
                   hipStream_t st)
{
  k_fill<<<1, TPB, 0, st>>>(acc, 0.f, 2L * F);
  tanh_acc(o0, kW, kb, acc,     N, F, st);
  tanh_acc(o1, kW, kb, acc + F, N, F, st);
  k_semantic<<<1, 32, 0, st>>>(acc, acc + F, q, 1.f / (float)N, F, attn);
  k_combine<<<cdiv(N * (long)F, TPB), TPB, 0, st>>>(o0, o1, attn, outp, N * (long)F);
}

extern "C" void kernel_launch(void* const* d_in, const int* in_sizes, int n_in,
                              void* d_out, int out_size, void* d_ws, size_t ws_size,
                              hipStream_t stream)
{
  // input order: x_author, x_paper, params1(13), params2(13), 6 edge arrays
  const float* x_a = (const float*)d_in[0];
  const float* x_p = (const float*)d_in[1];
  const long NA = in_sizes[0] / 64;
  const long NP = in_sizes[1] / 64;

  // params1 (din=64, dout=64, H=8, D=8)
  const float *Wa1 = (const float*)d_in[2],  *ba1 = (const float*)d_in[3];
  const float *Wp1 = (const float*)d_in[4],  *bp1 = (const float*)d_in[5];
  const float *aws1 = (const float*)d_in[6], *awd1 = (const float*)d_in[7];
  const float *ars1 = (const float*)d_in[8], *ard1 = (const float*)d_in[9];
  const float *acs1 = (const float*)d_in[10], *acd1 = (const float*)d_in[11];
  const float *kW1 = (const float*)d_in[12], *kb1 = (const float*)d_in[13];
  const float *q1  = (const float*)d_in[14];
  // params2 (din=64, dout=32, H=4, D=8)
  const float *Wa2 = (const float*)d_in[15], *ba2 = (const float*)d_in[16];
  const float *Wp2 = (const float*)d_in[17], *bp2 = (const float*)d_in[18];
  const float *aws2 = (const float*)d_in[19], *awd2 = (const float*)d_in[20];
  const float *ars2 = (const float*)d_in[21], *ard2 = (const float*)d_in[22];
  const float *acs2 = (const float*)d_in[23], *acd2 = (const float*)d_in[24];
  const float *kW2 = (const float*)d_in[25], *kb2 = (const float*)d_in[26];
  const float *q2  = (const float*)d_in[27];
  // edges
  const int *ws = (const int*)d_in[28], *wd = (const int*)d_in[29];
  const int *rs = (const int*)d_in[30], *rd = (const int*)d_in[31];
  const int *cs = (const int*)d_in[32], *cd = (const int*)d_in[33];
  const long Ew = in_sizes[28], Er = in_sizes[30], Ec = in_sizes[32];

  // workspace carve-out (~78M floats ~ 312MB); layer 2 reuses layer-1 buffers
  float* base = (float*)d_ws;
  size_t off = 0;
  auto alloc = [&](size_t n) { float* p = base + off; off += n; return p; };
  float* hA    = alloc(NA * 64);
  float* hP    = alloc(NP * 64);
  float* alSw  = alloc(NA * 8);  float* alDw = alloc(NP * 8);
  float* alSr  = alloc(NP * 8);  float* alDr = alloc(NA * 8);
  float* alSc  = alloc(NP * 8);  float* alDc = alloc(NP * 8);
  float* amaxw = alloc(NP * 8);  float* denw = alloc(NP * 8);
  float* amaxr = alloc(NA * 8);  float* denr = alloc(NA * 8);
  float* amaxc = alloc(NP * 8);  float* denc = alloc(NP * 8);
  float* ow    = alloc(NP * 64);
  float* orr   = alloc(NA * 64);
  float* oc    = alloc(NP * 64);
  float* outp1 = alloc(NP * 64);
  float* acc   = alloc(128);
  float* attn  = alloc(2);
  (void)ws_size; (void)n_in; (void)out_size;

  // ---------------- layer 1 (H=8, D=8, F=64) ----------------
  gemm(x_a, Wa1, ba1, hA, NA, 64, 64, stream);
  gemm(x_p, Wp1, bp1, hP, NP, 64, 64, stream);
  edge_out(hA, hP, aws1, awd1, ws, wd, Ew, NA, NP, 8, 8, alSw, alDw, amaxw, denw, ow,  stream);
  edge_out(hP, hA, ars1, ard1, rs, rd, Er, NP, NA, 8, 8, alSr, alDr, amaxr, denr, orr, stream);
  edge_out(hP, hP, acs1, acd1, cs, cd, Ec, NP, NP, 8, 8, alSc, alDc, amaxc, denc, oc,  stream);
  // authors: T=1 -> semantic attention is identity (out_a1 == orr)
  group2(ow, oc, kW1, kb1, q1, NP, 64, acc, attn, outp1, stream);
  k_relu<<<cdiv(NA * 64, TPB), TPB, 0, stream>>>(orr,   NA * 64);
  k_relu<<<cdiv(NP * 64, TPB), TPB, 0, stream>>>(outp1, NP * 64);

  // ---------------- layer 2 (H=4, D=8, F=32) ----------------
  float* outA = (float*)d_out;              // [NA,32]
  float* outP = (float*)d_out + NA * 32;    // [NP,32]
  gemm(orr,   Wa2, ba2, hA, NA, 64, 32, stream);
  gemm(outp1, Wp2, bp2, hP, NP, 64, 32, stream);
  edge_out(hA, hP, aws2, awd2, ws, wd, Ew, NA, NP, 4, 8, alSw, alDw, amaxw, denw, ow,   stream);
  edge_out(hP, hA, ars2, ard2, rs, rd, Er, NP, NA, 4, 8, alSr, alDr, amaxr, denr, outA, stream);
  edge_out(hP, hP, acs2, acd2, cs, cd, Ec, NP, NP, 4, 8, alSc, alDc, amaxc, denc, oc,   stream);
  group2(ow, oc, kW2, kb2, q2, NP, 32, acc, attn, outP, stream);
}
